// SchNet_39711267618821
// MI455X (gfx1250) — compile-verified, matrix operand order checked by reading, not run on hardware
//
#include <hip/hip_runtime.h>
#include <hip/hip_bf16.h>
#include <math.h>

typedef __attribute__((ext_vector_type(16))) _Float16 v16h;
typedef __attribute__((ext_vector_type(8)))  _Float16 h8;
typedef __attribute__((ext_vector_type(8)))  float    v8f;

#define PI_F 3.14159265358979323846f
#define CUTOFF_F 4.0f
#define NGAUSS 50
#define GPAD 64

__device__ __forceinline__ float sspf(float x) {
  // softplus(x) - ln2, overflow-safe
  float ax = fabsf(x);
  return fmaxf(x, 0.f) + log1pf(expf(-ax)) - 0.6931471805599453f;
}
__device__ __forceinline__ float sigmf(float x) { return 1.f / (1.f + expf(-x)); }

// ---------------------------------------------------------------------------
// WMMA f16->f32 GEMM: O = epilogue(A[M,KT] * B^T + bias), B stored [NC,KT]
// row-major (transposed weights pre-packed on host side of the launch).
//  KT (template): 64 or 128 -> K loop fully unrolled, A fragments hoisted.
//  ACT: 0 none | 1 ssp | 2 *= sigmoid(auxf32[idx]) | 3 *= sigmoid(auxf16[idx])
//  SCALE: *= scale[row]   (cosine cutoff per edge)
//  PRE:   store pre-activation (post-bias) to Opre as f16
//  OUT: 0 Oh=f16(v) | 1 Of=v | 2 Of+=v | 3 Of=resid[idx]+v | 4 Of=v, Oh=f16(ssp(v))
//  HASB:  bias present (compile-time)
// ---------------------------------------------------------------------------
template <int KT, int ACT, bool SCALE, bool PRE, int OUT, bool HASB>
__global__ __launch_bounds__(256) void gemm_k(
    const _Float16* __restrict__ A, const _Float16* __restrict__ B,
    const float* __restrict__ bias, const float* __restrict__ scale,
    const float* __restrict__ auxf, const _Float16* __restrict__ auxh,
    const float* __restrict__ resid,
    _Float16* __restrict__ Oh, float* __restrict__ Of, _Float16* __restrict__ Opre,
    int M, int NC) {
  constexpr int KS = KT / 32;                        // K steps
  const int lane = threadIdx.x & 31;
  const int wave = threadIdx.x >> 5;                 // 0..7
  const int m0 = blockIdx.x * 128 + wave * 16;       // this wave's 16 rows
  const int ml = lane & 15;
  const int hi = lane >> 4;                          // 0 or 1
  int rowA = m0 + ml;
  if (rowA >= M) rowA = M - 1;                       // clamp loads, guard stores
  const bool full = (m0 + 16) <= M;                  // wave-uniform store fast path
  const int nt = NC >> 4;

  // ---- load all A fragments once (reused across all n tiles) ----
  // layout (16-bit A 16x32): lanes0-15 K=k0+0..7,k0+16..23 ; lanes16-31 +8
  v16h afr[KS];
  const _Float16* arow = A + (size_t)rowA * KT;
#pragma unroll
  for (int kk = 0; kk < KS; ++kk) {
    const _Float16* ap = arow + kk * 32 + hi * 8;
    h8 a0 = *(const h8*)(ap);
    h8 a1 = *(const h8*)(ap + 16);
    afr[kk] = __builtin_shufflevector(a0, a1, 0, 1, 2, 3, 4, 5, 6, 7,
                                      8, 9, 10, 11, 12, 13, 14, 15);
  }

  for (int t = 0; t < nt; ++t) {
    const int n0 = t * 16;
    const _Float16* brow = B + (size_t)(n0 + ml) * KT;
    v8f acc = {0.f, 0.f, 0.f, 0.f, 0.f, 0.f, 0.f, 0.f};
#pragma unroll
    for (int kk = 0; kk < KS; ++kk) {
      const _Float16* bp = brow + kk * 32 + hi * 8;
      h8 b0 = *(const h8*)(bp);
      h8 b1 = *(const h8*)(bp + 16);
      v16h bf = __builtin_shufflevector(b0, b1, 0, 1, 2, 3, 4, 5, 6, 7,
                                        8, 9, 10, 11, 12, 13, 14, 15);
      acc = __builtin_amdgcn_wmma_f32_16x16x32_f16(false, afr[kk], false, bf,
                                                   (short)0, acc, false, false);
    }
    // C/D layout: VGPR r -> row m0 + 8*hi + r, col n0 + lane%16
    const int ocol = n0 + ml;
    auto epi = [&](int r) {
      const int orow = m0 + hi * 8 + r;
      const size_t idx = (size_t)orow * NC + ocol;
      float v = acc[r];
      if (HASB) v += bias[ocol];
      if (PRE) Opre[idx] = (_Float16)v;
      if (ACT == 1) v = sspf(v);
      else if (ACT == 2) v *= sigmf(auxf[idx]);
      else if (ACT == 3) v *= sigmf((float)auxh[idx]);
      if (SCALE) v *= scale[orow];
      if (OUT == 0) Oh[idx] = (_Float16)v;
      else if (OUT == 1) Of[idx] = v;
      else if (OUT == 2) Of[idx] += v;
      else if (OUT == 3) Of[idx] = resid[idx] + v;
      else { Of[idx] = v; Oh[idx] = (_Float16)sspf(v); }
    };
    if (full) {
#pragma unroll
      for (int r = 0; r < 8; ++r) epi(r);
    } else {
#pragma unroll
      for (int r = 0; r < 8; ++r)
        if (m0 + hi * 8 + r < M) epi(r);
    }
  }
}

// ---------------------------------------------------------------------------
// Elementwise / edge kernels
// ---------------------------------------------------------------------------
__global__ void zero_f32(float* p, size_t n) {
  size_t i = (size_t)blockIdx.x * blockDim.x + threadIdx.x;
  if (i < n) p[i] = 0.f;
}
__global__ void f32_to_f16(const float* __restrict__ s, _Float16* __restrict__ d, size_t n) {
  size_t i = (size_t)blockIdx.x * blockDim.x + threadIdx.x;
  if (i < n) d[i] = (_Float16)s[i];
}
// transpose-convert: s [L,R,C] f32 -> d [L,C,R] f16
__global__ void t_f32_to_f16(const float* __restrict__ s, _Float16* __restrict__ d,
                             int L, int R, int C) {
  size_t i = (size_t)blockIdx.x * blockDim.x + threadIdx.x;
  size_t tot = (size_t)L * R * C;
  if (i >= tot) return;
  int c = i % C;
  int r = (i / C) % R;
  int l = i / ((size_t)R * C);
  d[((size_t)l * C + c) * R + r] = (_Float16)s[i];
}
// mlp_w1 [L,50,F] -> padded f16 [L,GPAD,F]   (B' for backward g_ea GEMM)
__global__ void pad_w1_k(const float* __restrict__ s, _Float16* __restrict__ d,
                         int L, int F) {
  size_t i = (size_t)blockIdx.x * blockDim.x + threadIdx.x;
  size_t tot = (size_t)L * GPAD * F;
  if (i >= tot) return;
  int f = i % F;
  int g = (i / F) % GPAD;
  int l = i / ((size_t)F * GPAD);
  d[i] = (g < NGAUSS) ? (_Float16)s[((size_t)l * NGAUSS + g) * F + f] : (_Float16)0.f;
}
// mlp_w1 [L,50,F] -> padded transposed f16 [L,F,GPAD]  (B' for forward GEMM1)
__global__ void pad_w1t_k(const float* __restrict__ s, _Float16* __restrict__ d,
                          int L, int F) {
  size_t i = (size_t)blockIdx.x * blockDim.x + threadIdx.x;
  size_t tot = (size_t)L * F * GPAD;
  if (i >= tot) return;
  int g = i % GPAD;
  int f = (i / GPAD) % F;
  int l = i / ((size_t)F * GPAD);
  d[i] = (g < NGAUSS) ? (_Float16)s[((size_t)l * NGAUSS + g) * F + f] : (_Float16)0.f;
}
__global__ void gather_emb_k(const int* __restrict__ z, const float* __restrict__ emb,
                             float* __restrict__ h0, int N, int H) {
  size_t i = (size_t)blockIdx.x * blockDim.x + threadIdx.x;
  if (i >= (size_t)N * H) return;
  int n = i / H, f = i % H;
  h0[i] = emb[(size_t)z[n] * H + f];
}
// per-edge geometry + gaussian smearing + cutoff
__global__ void edge_geom_k(const float* __restrict__ pos, const int* __restrict__ ei,
                            const float* __restrict__ off, const float* __restrict__ cell,
                            float* __restrict__ vec, float* __restrict__ wv,
                            float* __restrict__ Cv, float* __restrict__ dCv,
                            _Float16* __restrict__ ea, int E) {
  int e = blockIdx.x * blockDim.x + threadIdx.x;
  if (e >= E) return;
  int r = ei[e], c = ei[E + e];
  float ox = off[e * 3 + 0], oy = off[e * 3 + 1], oz = off[e * 3 + 2];
  float vx = pos[r * 3 + 0] - pos[c * 3 + 0] + ox * cell[0] + oy * cell[3] + oz * cell[6];
  float vy = pos[r * 3 + 1] - pos[c * 3 + 1] + ox * cell[1] + oy * cell[4] + oz * cell[7];
  float vz = pos[r * 3 + 2] - pos[c * 3 + 2] + ox * cell[2] + oy * cell[5] + oz * cell[8];
  float w = sqrtf(vx * vx + vy * vy + vz * vz);
  vec[e * 3 + 0] = vx; vec[e * 3 + 1] = vy; vec[e * 3 + 2] = vz;
  wv[e] = w;
  float ph = w * (PI_F / CUTOFF_F);
  Cv[e] = 0.5f * (cosf(ph) + 1.f);
  dCv[e] = -0.5f * sinf(ph) * (PI_F / CUTOFF_F);
  const float dmu = CUTOFF_F / (NGAUSS - 1);
  const float coeff = -0.5f / (dmu * dmu);
  _Float16* er = ea + (size_t)e * GPAD;
#pragma unroll 4
  for (int g = 0; g < GPAD; ++g) {
    float d = w - (float)g * dmu;
    er[g] = (g < NGAUSS) ? (_Float16)expf(coeff * d * d) : (_Float16)0.f;
  }
}
// forward message + scatter: agg[col] += xh[row] * W
__global__ void edge_msg_k(const _Float16* __restrict__ xh, const _Float16* __restrict__ W,
                           const int* __restrict__ ei, float* __restrict__ agg,
                           int E, int F) {
  size_t i = (size_t)blockIdx.x * blockDim.x + threadIdx.x;
  if (i >= (size_t)E * F) return;
  int e = i / F, f = i % F;
  int r = ei[e], c = ei[E + e];
  atomicAdd(&agg[(size_t)c * F + f], (float)xh[(size_t)r * F + f] * (float)W[i]);
}
// backward edge: gW = gagg[col]*xh[row]; gu2 = gW*C; gxh[row]+=gagg[col]*u2*C;
// gw_c[e] += (sum_f gW*u2) * dC/dw
__global__ __launch_bounds__(128) void edge_bwd_k(
    const float* __restrict__ gagg, const _Float16* __restrict__ xh,
    const _Float16* __restrict__ u2, const float* __restrict__ Cv,
    const float* __restrict__ dCv, const int* __restrict__ ei,
    _Float16* __restrict__ gu2, float* __restrict__ gxh, float* __restrict__ gw_c,
    int E, int F) {
  int e = blockIdx.x;
  int f = threadIdx.x;
  int r = ei[e], c = ei[E + e];
  float ga = gagg[(size_t)c * F + f];
  float xv = (float)xh[(size_t)r * F + f];
  float u2v = (float)u2[(size_t)e * F + f];
  float Ce = Cv[e];
  float gW = ga * xv;
  gu2[(size_t)e * F + f] = (_Float16)(gW * Ce);
  atomicAdd(&gxh[(size_t)r * F + f], ga * u2v * Ce);
  __shared__ float red[128];
  red[f] = gW * u2v;
  __syncthreads();
  for (int s = 64; s > 0; s >>= 1) {
    if (f < s) red[f] += red[f + s];
    __syncthreads();
  }
  if (f == 0) gw_c[e] += red[0] * dCv[e];
}
// readout head: energy accumulation + d(ssp)/do1 * out2_w
__global__ void head_k(const float* __restrict__ o1, const float* __restrict__ w2,
                       const float* __restrict__ b2, float* __restrict__ energy,
                       _Float16* __restrict__ go1, int N, int Ho) {
  int n = blockIdx.x * blockDim.x + threadIdx.x;
  if (n >= N) return;
  float y = b2[0];
  for (int j = 0; j < Ho; ++j) {
    float x = o1[(size_t)n * Ho + j];
    y += sspf(x) * w2[j];
    go1[(size_t)n * Ho + j] = (_Float16)(sigmf(x) * w2[j]);
  }
  atomicAdd(energy, y);
}
// final: gw_total -> gvec -> force scatter
__global__ void edge_final_k(const float* __restrict__ g_ea, const _Float16* __restrict__ ea,
                             const float* __restrict__ wv, const float* __restrict__ vec,
                             const float* __restrict__ gw_c, const int* __restrict__ ei,
                             float* __restrict__ force, int E) {
  int e = blockIdx.x * blockDim.x + threadIdx.x;
  if (e >= E) return;
  float w = wv[e];
  float s = gw_c[e];
  const float dmu = CUTOFF_F / (NGAUSS - 1);
  const float coeff2 = -1.0f / (dmu * dmu);  // 2*coeff
  const float* ge = g_ea + (size_t)e * GPAD;
  const _Float16* er = ea + (size_t)e * GPAD;
#pragma unroll 5
  for (int g = 0; g < NGAUSS; ++g)
    s += ge[g] * (float)er[g] * coeff2 * (w - (float)g * dmu);
  float inv = (w > 1e-12f) ? (s / w) : 0.f;
  float gx = inv * vec[e * 3 + 0], gy = inv * vec[e * 3 + 1], gz = inv * vec[e * 3 + 2];
  int r = ei[e], c = ei[E + e];
  atomicAdd(&force[r * 3 + 0], -gx); atomicAdd(&force[r * 3 + 1], -gy); atomicAdd(&force[r * 3 + 2], -gz);
  atomicAdd(&force[c * 3 + 0],  gx); atomicAdd(&force[c * 3 + 1],  gy); atomicAdd(&force[c * 3 + 2],  gz);
}

// ---------------------------------------------------------------------------
static inline dim3 gblk(size_t n, int b) { return dim3((unsigned)((n + b - 1) / b)); }

extern "C" void kernel_launch(void* const* d_in, const int* in_sizes, int n_in,
                              void* d_out, int out_size, void* d_ws, size_t ws_size,
                              hipStream_t stream) {
  const int N = in_sizes[0];
  const int E = in_sizes[2] / 2;
  const int H = in_sizes[5] / 100;     // 128
  const int F = H;                     // 128
  const int L = in_sizes[7] / F;       // 6
  const int Ho = H / 2;                // 64

  const int*   z       = (const int*)d_in[0];
  const float* pos     = (const float*)d_in[1];
  const int*   ei      = (const int*)d_in[2];
  const float* offset  = (const float*)d_in[3];
  const float* cell    = (const float*)d_in[4];
  const float* emb     = (const float*)d_in[5];
  const float* mlp_w1  = (const float*)d_in[6];
  const float* mlp_b1  = (const float*)d_in[7];
  const float* mlp_w2  = (const float*)d_in[8];
  const float* mlp_b2  = (const float*)d_in[9];
  const float* conv_w1 = (const float*)d_in[10];
  const float* conv_w2 = (const float*)d_in[11];
  const float* conv_b2 = (const float*)d_in[12];
  const float* lin_w   = (const float*)d_in[13];
  const float* lin_b   = (const float*)d_in[14];
  const float* out1_w  = (const float*)d_in[15];
  const float* out1_b  = (const float*)d_in[16];
  const float* out2_w  = (const float*)d_in[17];
  const float* out2_b  = (const float*)d_in[18];

  float* energy = (float*)d_out;
  float* force  = energy + 1;

  // ---- workspace carve-out ----
  char* p = (char*)d_ws;
  auto carve = [&](size_t bytes) -> char* {
    char* r = p;
    p += (bytes + 255) & ~(size_t)255;
    return r;
  };
  // weights: both orientations, f16, so every GEMM loads B as [NC,K] contiguous
  _Float16* w1p16   = (_Float16*)carve((size_t)L * GPAD * F * 2);  // [GPAD,F]  (bwd g_ea)
  _Float16* w1t16   = (_Float16*)carve((size_t)L * F * GPAD * 2);  // [F,GPAD]  (fwd GEMM1)
  _Float16* w2_16   = (_Float16*)carve((size_t)L * F * F * 2);     // orig      (bwd gu1)
  _Float16* w2t16   = (_Float16*)carve((size_t)L * F * F * 2);     // transpose (fwd GEMM2/u2)
  _Float16* cw1_16  = (_Float16*)carve((size_t)L * H * F * 2);     // orig      (bwd gh)
  _Float16* cw1t16  = (_Float16*)carve((size_t)L * F * H * 2);     // transpose (xh)
  _Float16* cw2_16  = (_Float16*)carve((size_t)L * F * H * 2);     // orig      (bwd gagg)
  _Float16* cw2t16  = (_Float16*)carve((size_t)L * H * F * 2);     // transpose (m)
  _Float16* lw_16   = (_Float16*)carve((size_t)L * H * H * 2);     // orig      (bwd gm)
  _Float16* lwt16   = (_Float16*)carve((size_t)L * H * H * 2);     // transpose (h_out)
  _Float16* o1w16   = (_Float16*)carve((size_t)H * Ho * 2);        // orig      (gh head)
  _Float16* o1wt16  = (_Float16*)carve((size_t)Ho * H * 2);        // transpose (o1)
  _Float16* ea16    = (_Float16*)carve((size_t)E * GPAD * 2);
  float*    vec     = (float*)carve((size_t)E * 3 * 4);
  float*    w_e     = (float*)carve((size_t)E * 4);
  float*    C_e     = (float*)carve((size_t)E * 4);
  float*    dC_e    = (float*)carve((size_t)E * 4);
  float*    h_all   = (float*)carve((size_t)(L + 1) * N * H * 4);
  float*    m_all   = (float*)carve((size_t)L * N * H * 4);
  _Float16* h16     = (_Float16*)carve((size_t)N * H * 2);
  _Float16* xh16    = (_Float16*)carve((size_t)N * F * 2);
  float*    agg     = (float*)carve((size_t)N * F * 4);
  _Float16* agg16   = (_Float16*)carve((size_t)N * F * 2);
  _Float16* s16     = (_Float16*)carve((size_t)N * H * 2);
  _Float16* gm16    = (_Float16*)carve((size_t)N * H * 2);
  float*    gagg    = (float*)carve((size_t)N * F * 4);
  float*    gxh     = (float*)carve((size_t)N * F * 4);
  _Float16* gxh16   = (_Float16*)carve((size_t)N * F * 2);
  float*    gh      = (float*)carve((size_t)N * H * 4);
  _Float16* gh16    = (_Float16*)carve((size_t)N * H * 2);
  float*    o1      = (float*)carve((size_t)N * Ho * 4);
  _Float16* go1     = (_Float16*)carve((size_t)N * Ho * 2);
  _Float16* bufA    = (_Float16*)carve((size_t)E * F * 2);  // u1
  _Float16* bufB    = (_Float16*)carve((size_t)E * F * 2);  // a1 -> gu2
  _Float16* bufC    = (_Float16*)carve((size_t)E * F * 2);  // W / u2 -> gu1
  float*    g_ea    = (float*)carve((size_t)E * GPAD * 4);
  float*    gw_c    = (float*)carve((size_t)E * 4);
  (void)ws_size; (void)n_in;

  const _Float16* H16NULL = nullptr;
  const float*    F32NULL = nullptr;

#define GEMM(KT, ACT, SC, PRE, OUTM, HASB, A_, B_, bias_, scale_, auxf_, auxh_, resid_, Oh_, Of_, Op_, M_, NC_) \
  gemm_k<KT, ACT, SC, PRE, OUTM, HASB><<<gblk((size_t)(M_), 128), dim3(256), 0, stream>>>(                      \
      A_, B_, bias_, scale_, auxf_, auxh_, resid_, Oh_, Of_, Op_, M_, NC_)

  // ---- init / weight conversion ----
  zero_f32<<<gblk(out_size, 256), 256, 0, stream>>>(energy, (size_t)out_size);
  zero_f32<<<gblk((size_t)E * GPAD, 256), 256, 0, stream>>>(g_ea, (size_t)E * GPAD);
  zero_f32<<<gblk(E, 256), 256, 0, stream>>>(gw_c, (size_t)E);
  pad_w1_k<<<gblk((size_t)L * GPAD * F, 256), 256, 0, stream>>>(mlp_w1, w1p16, L, F);
  pad_w1t_k<<<gblk((size_t)L * F * GPAD, 256), 256, 0, stream>>>(mlp_w1, w1t16, L, F);
  f32_to_f16<<<gblk((size_t)L * F * F, 256), 256, 0, stream>>>(mlp_w2, w2_16, (size_t)L * F * F);
  t_f32_to_f16<<<gblk((size_t)L * F * F, 256), 256, 0, stream>>>(mlp_w2, w2t16, L, F, F);
  f32_to_f16<<<gblk((size_t)L * H * F, 256), 256, 0, stream>>>(conv_w1, cw1_16, (size_t)L * H * F);
  t_f32_to_f16<<<gblk((size_t)L * H * F, 256), 256, 0, stream>>>(conv_w1, cw1t16, L, H, F);
  f32_to_f16<<<gblk((size_t)L * F * H, 256), 256, 0, stream>>>(conv_w2, cw2_16, (size_t)L * F * H);
  t_f32_to_f16<<<gblk((size_t)L * F * H, 256), 256, 0, stream>>>(conv_w2, cw2t16, L, F, H);
  f32_to_f16<<<gblk((size_t)L * H * H, 256), 256, 0, stream>>>(lin_w, lw_16, (size_t)L * H * H);
  t_f32_to_f16<<<gblk((size_t)L * H * H, 256), 256, 0, stream>>>(lin_w, lwt16, L, H, H);
  f32_to_f16<<<gblk((size_t)H * Ho, 256), 256, 0, stream>>>(out1_w, o1w16, (size_t)H * Ho);
  t_f32_to_f16<<<gblk((size_t)H * Ho, 256), 256, 0, stream>>>(out1_w, o1wt16, 1, H, Ho);

  edge_geom_k<<<gblk(E, 256), 256, 0, stream>>>(pos, ei, offset, cell, vec, w_e, C_e, dC_e, ea16, E);
  gather_emb_k<<<gblk((size_t)N * H, 256), 256, 0, stream>>>(z, emb, h_all, N, H);

  // ================= forward =================
  for (int i = 0; i < L; ++i) {
    float* hcur = h_all + (size_t)i * N * H;
    float* hnxt = h_all + (size_t)(i + 1) * N * H;
    float* mi   = m_all + (size_t)i * N * H;
    // a1 = ssp(ea @ w1 + b1)          (B' = w1t [F,GPAD])
    GEMM(GPAD, 1, false, false, 0, true, ea16, w1t16 + (size_t)i * F * GPAD, mlp_b1 + (size_t)i * F,
         F32NULL, F32NULL, H16NULL, F32NULL, bufB, (float*)nullptr, (_Float16*)nullptr, E, F);
    // W = (a1 @ w2 + b2) * C          (B' = w2t [F,F])
    GEMM(128, 0, true, false, 0, true, bufB, w2t16 + (size_t)i * F * F, mlp_b2 + (size_t)i * F,
         C_e, F32NULL, H16NULL, F32NULL, bufC, (float*)nullptr, (_Float16*)nullptr, E, F);
    // xh = h @ cw1                    (B' = cw1t [F,H])
    f32_to_f16<<<gblk((size_t)N * H, 256), 256, 0, stream>>>(hcur, h16, (size_t)N * H);
    GEMM(128, 0, false, false, 0, false, h16, cw1t16 + (size_t)i * F * H, F32NULL,
         F32NULL, F32NULL, H16NULL, F32NULL, xh16, (float*)nullptr, (_Float16*)nullptr, N, F);
    // agg = segment_sum(xh[row]*W, col)
    zero_f32<<<gblk((size_t)N * F, 256), 256, 0, stream>>>(agg, (size_t)N * F);
    edge_msg_k<<<gblk((size_t)E * F, 256), 256, 0, stream>>>(xh16, bufC, ei, agg, E, F);
    // m = agg @ cw2 + cb2 (store f32) ; s = ssp(m) f16   (B' = cw2t [H,F])
    f32_to_f16<<<gblk((size_t)N * F, 256), 256, 0, stream>>>(agg, agg16, (size_t)N * F);
    GEMM(128, 0, false, false, 4, true, agg16, cw2t16 + (size_t)i * H * F, conv_b2 + (size_t)i * H,
         F32NULL, F32NULL, H16NULL, F32NULL, s16, mi, (_Float16*)nullptr, N, H);
    // h_next = h + s @ lw + lb        (B' = lwt [H,H])
    GEMM(128, 0, false, false, 3, true, s16, lwt16 + (size_t)i * H * H, lin_b + (size_t)i * H,
         F32NULL, F32NULL, H16NULL, hcur, (_Float16*)nullptr, hnxt, (_Float16*)nullptr, N, H);
  }

  // ================= readout head (fwd + seed of backward) =================
  {
    float* hfin = h_all + (size_t)L * N * H;
    f32_to_f16<<<gblk((size_t)N * H, 256), 256, 0, stream>>>(hfin, h16, (size_t)N * H);
    // o1 = h @ out1_w + b             (B' = o1wt [Ho,H])
    GEMM(128, 0, false, false, 1, true, h16, o1wt16, out1_b,
         F32NULL, F32NULL, H16NULL, F32NULL, (_Float16*)nullptr, o1, (_Float16*)nullptr, N, Ho);
    head_k<<<gblk(N, 256), 256, 0, stream>>>(o1, out2_w, out2_b, energy, go1, N, Ho);
    // gh = go1 @ out1_w^T             (B' = o1w [H,Ho])
    GEMM(64, 0, false, false, 1, false, go1, o1w16, F32NULL,
         F32NULL, F32NULL, H16NULL, F32NULL, (_Float16*)nullptr, gh, (_Float16*)nullptr, N, H);
  }

  // ================= backward =================
  for (int i = L - 1; i >= 0; --i) {
    float* hcur = h_all + (size_t)i * N * H;
    float* mi   = m_all + (size_t)i * N * H;
    // recompute xh
    f32_to_f16<<<gblk((size_t)N * H, 256), 256, 0, stream>>>(hcur, h16, (size_t)N * H);
    GEMM(128, 0, false, false, 0, false, h16, cw1t16 + (size_t)i * F * H, F32NULL,
         F32NULL, F32NULL, H16NULL, F32NULL, xh16, (float*)nullptr, (_Float16*)nullptr, N, F);
    // recompute u1 (pre) + a1 (post)
    GEMM(GPAD, 1, false, true, 0, true, ea16, w1t16 + (size_t)i * F * GPAD, mlp_b1 + (size_t)i * F,
         F32NULL, F32NULL, H16NULL, F32NULL, bufB, (float*)nullptr, bufA, E, F);
    // recompute u2 = a1 @ w2 + b2
    GEMM(128, 0, false, false, 0, true, bufB, w2t16 + (size_t)i * F * F, mlp_b2 + (size_t)i * F,
         F32NULL, F32NULL, H16NULL, F32NULL, bufC, (float*)nullptr, (_Float16*)nullptr, E, F);
    // gm = (gh @ lw^T) * sigmoid(m)   (B' = lw orig [H,H])
    f32_to_f16<<<gblk((size_t)N * H, 256), 256, 0, stream>>>(gh, gh16, (size_t)N * H);
    GEMM(128, 2, false, false, 0, false, gh16, lw_16 + (size_t)i * H * H, F32NULL,
         F32NULL, mi, H16NULL, F32NULL, gm16, (float*)nullptr, (_Float16*)nullptr, N, H);
    // gagg = gm @ cw2^T               (B' = cw2 orig [F,H])
    GEMM(128, 0, false, false, 1, false, gm16, cw2_16 + (size_t)i * F * H, F32NULL,
         F32NULL, F32NULL, H16NULL, F32NULL, (_Float16*)nullptr, gagg, (_Float16*)nullptr, N, F);
    // edge backward: gu2 (into bufB), gxh scatter, gw_c += gC*dC/dw
    zero_f32<<<gblk((size_t)N * F, 256), 256, 0, stream>>>(gxh, (size_t)N * F);
    edge_bwd_k<<<dim3((unsigned)E), dim3(128), 0, stream>>>(gagg, xh16, bufC, C_e, dC_e, ei,
                                                            bufB, gxh, gw_c, E, F);
    // gu1 = (gu2 @ w2^T) * sigmoid(u1)  (B' = w2 orig [F,F]) -> bufC
    GEMM(128, 3, false, false, 0, false, bufB, w2_16 + (size_t)i * F * F, F32NULL,
         F32NULL, F32NULL, bufA, F32NULL, bufC, (float*)nullptr, (_Float16*)nullptr, E, F);
    // g_ea += gu1 @ w1^T              (B' = w1p [GPAD,F])
    GEMM(128, 0, false, false, 2, false, bufC, w1p16 + (size_t)i * GPAD * F, F32NULL,
         F32NULL, F32NULL, H16NULL, F32NULL, (_Float16*)nullptr, g_ea, (_Float16*)nullptr, E, GPAD);
    // gh = gh + gxh @ cw1^T           (B' = cw1 orig [H,F])
    f32_to_f16<<<gblk((size_t)N * F, 256), 256, 0, stream>>>(gxh, gxh16, (size_t)N * F);
    GEMM(128, 0, false, false, 3, false, gxh16, cw1_16 + (size_t)i * H * F, F32NULL,
         F32NULL, F32NULL, H16NULL, gh, (_Float16*)nullptr, gh, (_Float16*)nullptr, N, H);
  }

  // ================= distance grad -> forces =================
  edge_final_k<<<gblk(E, 256), 256, 0, stream>>>(g_ea, ea16, w_e, vec, gw_c, ei, force, E);
#undef GEMM
}
